// WaveConv1d_37967510897048
// MI455X (gfx1250) — compile-verified, matrix operand order checked by reading, not run on hardware
//
#include <hip/hip_runtime.h>
#include <hip/hip_bf16.h>

typedef __attribute__((ext_vector_type(16))) _Float16 v16h;
typedef __attribute__((ext_vector_type(8)))  float    v8f;
typedef __attribute__((ext_vector_type(4)))  float    f4;
typedef __attribute__((ext_vector_type(4)))  _Float16 v4h;
typedef __attribute__((ext_vector_type(2)))  __fp16   fp16x2;

union Frag16 { v16h v; unsigned int u[8]; };
union FragC  { v8f  v; float f[8]; };

#define ACT_NONE 0
#define ACT_GELU 1
#define ACT_SOFTPLUS 2

#define BM 64
#define BN 64
#define BK 32

__device__ inline v4h cvt4(f4 t) {
    union { fp16x2 p[2]; v4h h; } r;
    r.p[0] = __builtin_amdgcn_cvt_pkrtz(t.x, t.y);
    r.p[1] = __builtin_amdgcn_cvt_pkrtz(t.z, t.w);
    return r.h;
}

// ---------------------------------------------------------------------------
// 4-level Haar DWT, one block per (b,c) row of length 1024
// ---------------------------------------------------------------------------
__global__ __launch_bounds__(512) void dwt_kernel(
    const float* __restrict__ x,
    float* __restrict__ yh0, float* __restrict__ yh1,
    float* __restrict__ yh2, float* __restrict__ yh3,
    float* __restrict__ yl)
{
    __shared__ float buf[1024];
    const float IS = 0.70710678118654752f;
    int row = blockIdx.x;            // b*256 + c
    int tid = threadIdx.x;
    buf[tid]       = x[(size_t)row * 1024 + tid];
    buf[tid + 512] = x[(size_t)row * 1024 + tid + 512];
    __syncthreads();
    float* yhp[4] = { yh0, yh1, yh2, yh3 };
    int    Ls[4]  = { 512, 256, 128, 64 };
    for (int l = 0; l < 4; ++l) {
        int hl = Ls[l];
        float a = 0.0f;
        if (tid < hl) {
            float xe = buf[2 * tid], xo = buf[2 * tid + 1];
            a = (xe + xo) * IS;
            yhp[l][(size_t)row * hl + tid] = (xe - xo) * IS;
        }
        __syncthreads();
        if (tid < hl) buf[tid] = a;
        __syncthreads();
    }
    if (tid < 64) yl[(size_t)row * 64 + tid] = buf[tid];
}

// ---------------------------------------------------------------------------
// einsum('bix,iox->box'): per-position 16x256 @ 256x256 matmul.
// Block = 16 waves; wave w owns position x0+w; o-tile of 16; K=256 loop.
// Weights streamed coalesced along x (contiguous axis) as b128 loads.
// ---------------------------------------------------------------------------
__global__ __launch_bounds__(512) void bmm_pos_kernel(
    const float* __restrict__ In, int inLayout, // 0: (b,i,x)  1: (b,x,i)
    const float* __restrict__ W,                // (i, 256, L)
    float* __restrict__ Out,                    // (b, 256, L)
    int L)
{
    __shared__ __align__(16) _Float16 Ast[16][16][32]; // [x][b][k]
    __shared__ __align__(16) _Float16 Bst[16][16][32]; // [x][o][k]
    int tid  = threadIdx.x;
    int lane = tid & 31;
    int wave = tid >> 5;                 // x within chunk
    int hf   = lane >> 4;
    int l16  = lane & 15;
    int x0   = blockIdx.x * 16;
    int o0   = blockIdx.y * 16;

    v8f acc = {0,0,0,0,0,0,0,0};

    for (int i0 = 0; i0 < 256; i0 += 32) {
        f4 ta[4], tb[4];
        if (inLayout == 0) {
            #pragma unroll
            for (int e = 0; e < 4; ++e) {           // 2048 float4, 512 thr
                int idx = tid + e * 512;
                int xq = idx & 3, bb = (idx >> 2) & 15, k = idx >> 6;
                ta[e] = *(const f4*)&In[((size_t)bb * 256 + i0 + k) * L + x0 + xq * 4];
            }
        } else {
            #pragma unroll
            for (int e = 0; e < 4; ++e) {
                int idx = tid + e * 512;
                int kq = idx & 7, xx = (idx >> 3) & 15, bb = idx >> 7;
                ta[e] = *(const f4*)&In[((size_t)bb * L + x0 + xx) * 256 + i0 + kq * 4];
            }
        }
        #pragma unroll
        for (int e = 0; e < 4; ++e) {
            int idx = tid + e * 512;
            int xq = idx & 3, oo = (idx >> 2) & 15, k = idx >> 6;
            tb[e] = *(const f4*)&W[((size_t)(i0 + k) * 256 + o0 + oo) * L + x0 + xq * 4];
        }
        if (inLayout == 0) {
            #pragma unroll
            for (int e = 0; e < 4; ++e) {
                int idx = tid + e * 512;
                int xq = idx & 3, bb = (idx >> 2) & 15, k = idx >> 6;
                v4h h = cvt4(ta[e]);
                Ast[xq * 4 + 0][bb][k] = h.x;
                Ast[xq * 4 + 1][bb][k] = h.y;
                Ast[xq * 4 + 2][bb][k] = h.z;
                Ast[xq * 4 + 3][bb][k] = h.w;
            }
        } else {
            #pragma unroll
            for (int e = 0; e < 4; ++e) {
                int idx = tid + e * 512;
                int kq = idx & 7, xx = (idx >> 3) & 15, bb = idx >> 7;
                *(v4h*)&Ast[xx][bb][kq * 4] = cvt4(ta[e]);
            }
        }
        #pragma unroll
        for (int e = 0; e < 4; ++e) {
            int idx = tid + e * 512;
            int xq = idx & 3, oo = (idx >> 2) & 15, k = idx >> 6;
            v4h h = cvt4(tb[e]);
            Bst[xq * 4 + 0][oo][k] = h.x;
            Bst[xq * 4 + 1][oo][k] = h.y;
            Bst[xq * 4 + 2][oo][k] = h.z;
            Bst[xq * 4 + 3][oo][k] = h.w;
        }
        __syncthreads();

        Frag16 af, bf;
        #pragma unroll
        for (int q = 0; q < 4; ++q) {
            int k1 = hf * 8 + 2 * q;
            int k2 = 16 + hf * 8 + 2 * q;
            af.u[q]     = *(const unsigned int*)&Ast[wave][l16][k1];
            af.u[q + 4] = *(const unsigned int*)&Ast[wave][l16][k2];
            bf.u[q]     = *(const unsigned int*)&Bst[wave][l16][k1];
            bf.u[q + 4] = *(const unsigned int*)&Bst[wave][l16][k2];
        }
        acc = __builtin_amdgcn_wmma_f32_16x16x32_f16(
            false, af.v, false, bf.v, (short)0, acc, false, false);
        __syncthreads();
    }

    FragC fc; fc.v = acc;
    #pragma unroll
    for (int q = 0; q < 8; ++q) {
        int bb = q + 8 * hf;             // batch row
        Out[((size_t)bb * 256 + o0 + l16) * L + x0 + wave] = fc.f[q];
    }
}

// ---------------------------------------------------------------------------
// Generic WMMA GEMM: C = act(op(A) @ B^T + bias), B stored [N x K] (ldb)
//   aMode 0: A row-major [M x K], lda = aParam
//   aMode 1: A is (b, K, x) with r = b*Lx + x, Lx = aParam (yh^T access)
// Tile 64x64x32, 256 threads (8 waves), 2 C-subtiles per wave.
// All K and leading dims are multiples of 4 -> b128 staging loads.
// ---------------------------------------------------------------------------
__global__ __launch_bounds__(256) void gemm_wmma_kernel(
    const float* __restrict__ A, int aMode, int aParam,
    const float* __restrict__ B, int ldb,
    const float* __restrict__ bias,
    float* __restrict__ C, int ldc,
    int M, int N, int K, int act)
{
    __shared__ __align__(16) _Float16 As[BM][BK];
    __shared__ __align__(16) _Float16 Bs[BN][BK];   // [n][k]
    int tid  = threadIdx.x;
    int lane = tid & 31;
    int wave = tid >> 5;
    int hf   = lane >> 4;
    int l16  = lane & 15;
    int gm0  = blockIdx.y * BM;
    int gn0  = blockIdx.x * BN;
    int ncol  = wave & 3;
    int mrow0 = (wave >> 2) * 2;

    v8f acc0 = {0,0,0,0,0,0,0,0};
    v8f acc1 = {0,0,0,0,0,0,0,0};

    for (int k0 = 0; k0 < K; k0 += BK) {
        f4 ta[2], tb[2];
        const f4 fz = {0.0f, 0.0f, 0.0f, 0.0f};
        if (aMode == 0) {
            #pragma unroll
            for (int e = 0; e < 2; ++e) {           // 512 float4, 256 thr
                int idx = tid + e * 256;
                int kq = idx & 7, r = idx >> 3;
                int gr = gm0 + r, gk = k0 + kq * 4;
                ta[e] = (gr < M && gk < K) ? *(const f4*)&A[(size_t)gr * aParam + gk] : fz;
            }
        } else {
            // contiguous along r (x axis); M = 16L, blocks never straddle b
            #pragma unroll
            for (int e = 0; e < 2; ++e) {
                int idx = tid + e * 256;
                int rq = idx & 15, k = idx >> 4;
                int gr = gm0 + rq * 4, gk = k0 + k;
                if (gk < K) {
                    int bb = gr / aParam, xx = gr % aParam;
                    ta[e] = *(const f4*)&A[(size_t)bb * K * aParam + (size_t)gk * aParam + xx];
                } else ta[e] = fz;
            }
        }
        #pragma unroll
        for (int e = 0; e < 2; ++e) {
            int idx = tid + e * 256;
            int kq = idx & 7, n = idx >> 3;
            int gn = gn0 + n, gk = k0 + kq * 4;
            tb[e] = (gn < N && gk < K) ? *(const f4*)&B[(size_t)gn * ldb + gk] : fz;
        }
        if (aMode == 0) {
            #pragma unroll
            for (int e = 0; e < 2; ++e) {
                int idx = tid + e * 256;
                int kq = idx & 7, r = idx >> 3;
                *(v4h*)&As[r][kq * 4] = cvt4(ta[e]);
            }
        } else {
            #pragma unroll
            for (int e = 0; e < 2; ++e) {
                int idx = tid + e * 256;
                int rq = idx & 15, k = idx >> 4;
                v4h h = cvt4(ta[e]);
                As[rq * 4 + 0][k] = h.x;
                As[rq * 4 + 1][k] = h.y;
                As[rq * 4 + 2][k] = h.z;
                As[rq * 4 + 3][k] = h.w;
            }
        }
        #pragma unroll
        for (int e = 0; e < 2; ++e) {
            int idx = tid + e * 256;
            int kq = idx & 7, n = idx >> 3;
            *(v4h*)&Bs[n][kq * 4] = cvt4(tb[e]);
        }
        __syncthreads();

        Frag16 bfr;
        {
            int n = ncol * 16 + l16;
            #pragma unroll
            for (int q = 0; q < 4; ++q) {
                int k1 = hf * 8 + 2 * q;
                int k2 = 16 + hf * 8 + 2 * q;
                bfr.u[q]     = *(const unsigned int*)&Bs[n][k1];
                bfr.u[q + 4] = *(const unsigned int*)&Bs[n][k2];
            }
        }
        #pragma unroll
        for (int j = 0; j < 2; ++j) {
            Frag16 af;
            int r = (mrow0 + j) * 16 + l16;
            #pragma unroll
            for (int q = 0; q < 4; ++q) {
                int k1 = hf * 8 + 2 * q;
                int k2 = 16 + hf * 8 + 2 * q;
                af.u[q]     = *(const unsigned int*)&As[r][k1];
                af.u[q + 4] = *(const unsigned int*)&As[r][k2];
            }
            if (j == 0)
                acc0 = __builtin_amdgcn_wmma_f32_16x16x32_f16(
                    false, af.v, false, bfr.v, (short)0, acc0, false, false);
            else
                acc1 = __builtin_amdgcn_wmma_f32_16x16x32_f16(
                    false, af.v, false, bfr.v, (short)0, acc1, false, false);
        }
        __syncthreads();
    }

    int gn = gn0 + ncol * 16 + l16;
    float bv = (bias != nullptr && gn < N) ? bias[gn] : 0.0f;
    #pragma unroll
    for (int j = 0; j < 2; ++j) {
        FragC fc; fc.v = j ? acc1 : acc0;
        #pragma unroll
        for (int q = 0; q < 8; ++q) {
            int gm = gm0 + (mrow0 + j) * 16 + q + 8 * hf;
            if (gm < M && gn < N) {
                float val = fc.f[q] + bv;
                if (act == ACT_GELU)
                    val = 0.5f * val * (1.0f + erff(val * 0.70710678f));
                else if (act == ACT_SOFTPLUS)
                    val = (val > 20.0f) ? val : log1pf(__expf(val));
                C[(size_t)gm * ldc + gn] = val;
            }
        }
    }
}

// ---------------------------------------------------------------------------
// Depthwise causal conv (width 2 over s) + SiLU, reading xh slice of xz
// ---------------------------------------------------------------------------
__global__ void conv_silu_kernel(
    const float* __restrict__ xz, const float* __restrict__ cw,
    const float* __restrict__ cb, float* __restrict__ xhc, int L)
{
    int idx = blockIdx.x * blockDim.x + threadIdx.x;
    if (idx >= 16 * 256 * L) return;
    int d = idx % L;
    int s = (idx / L) % 256;
    int b = idx / (256 * L);
    size_t rb = ((size_t)b * 256 + s) * 2 * L;
    float cur  = xz[rb + d];
    float prev = (s > 0) ? xz[rb - 2 * (size_t)L + d] : 0.0f;
    float v = prev * cw[2 * d] + cur * cw[2 * d + 1] + cb[d];
    v = v / (1.0f + __expf(-v));
    xhc[((size_t)b * 256 + s) * L + d] = v;
}

// ---------------------------------------------------------------------------
// Selective-scan over s=256; thread per (b,d); 16 states in registers.
// Fuses y = (scan + D*xh) * silu(z).
// ---------------------------------------------------------------------------
__global__ void scan_kernel(
    const float* __restrict__ dtf, const float* __restrict__ xhc,
    const float* __restrict__ dbl, const float* __restrict__ xz,
    const float* __restrict__ alog, const float* __restrict__ Dp,
    float* __restrict__ y2, int L, int R)
{
    int idx = blockIdx.x * blockDim.x + threadIdx.x;
    if (idx >= 16 * L) return;
    int b = idx / L, d = idx % L;
    float Arow[16], h[16];
    #pragma unroll
    for (int n = 0; n < 16; ++n) {
        Arow[n] = -__expf(alog[d * 16 + n]);
        h[n] = 0.0f;
    }
    int   E  = R + 32;
    float Dd = Dp[d];
    for (int s = 0; s < 256; ++s) {
        size_t base = (size_t)b * 256 + s;
        float dtv = dtf[base * L + d];
        float xv  = xhc[base * L + d];
        float zv  = xz[base * 2 * L + L + d];
        const float* BC = dbl + base * E + R;
        float y = 0.0f;
        #pragma unroll
        for (int n = 0; n < 16; ++n) {
            float dA = __expf(dtv * Arow[n]);
            h[n] = dA * h[n] + dtv * BC[n] * xv;
            y += h[n] * BC[16 + n];
        }
        y2[base * L + d] = (y + Dd * xv) * (zv / (1.0f + __expf(-zv)));
    }
}

// ---------------------------------------------------------------------------
// res += mamba_out; LayerNorm over the 256-channel axis; wave per (b,x);
// writes result in place into res (becomes new_yh).
// ---------------------------------------------------------------------------
__global__ __launch_bounds__(256) void ln_res_kernel(
    float* __restrict__ res, const float* __restrict__ mo,
    const float* __restrict__ g, const float* __restrict__ beta, int L)
{
    int tid  = threadIdx.x;
    int lane = tid & 31;
    int wid  = blockIdx.x * 8 + (tid >> 5);
    if (wid >= 16 * L) return;
    int b = wid / L, x = wid % L;
    float vals[8];
    float sum = 0.0f;
    #pragma unroll
    for (int j = 0; j < 8; ++j) {
        int c = j * 32 + lane;
        size_t a = ((size_t)b * 256 + c) * L + x;
        vals[j] = res[a] + mo[a];
        sum += vals[j];
    }
    #pragma unroll
    for (int off = 16; off > 0; off >>= 1) sum += __shfl_xor(sum, off, 32);
    float mean = sum * (1.0f / 256.0f);
    float vs = 0.0f;
    #pragma unroll
    for (int j = 0; j < 8; ++j) { float dd = vals[j] - mean; vs += dd * dd; }
    #pragma unroll
    for (int off = 16; off > 0; off >>= 1) vs += __shfl_xor(vs, off, 32);
    float rstd = rsqrtf(vs * (1.0f / 256.0f) + 1e-5f);
    #pragma unroll
    for (int j = 0; j < 8; ++j) {
        int c = j * 32 + lane;
        size_t a = ((size_t)b * 256 + c) * L + x;
        res[a] = (vals[j] - mean) * rstd * g[c] + beta[c];
    }
}

// ---------------------------------------------------------------------------
// Haar inverse DWT: (ya,yd) of length L -> out of length 2L
// ---------------------------------------------------------------------------
__global__ void idwt_kernel(
    const float* __restrict__ ya, const float* __restrict__ yd,
    float* __restrict__ out, int L)
{
    int idx = blockIdx.x * blockDim.x + threadIdx.x;
    if (idx >= 16 * 256 * L) return;
    const float IS = 0.70710678118654752f;
    int t = idx % L, row = idx / L;
    float a = ya[(size_t)row * L + t], d = yd[(size_t)row * L + t];
    out[(size_t)row * 2 * L + 2 * t]     = (a + d) * IS;
    out[(size_t)row * 2 * L + 2 * t + 1] = (a - d) * IS;
}

// ---------------------------------------------------------------------------
extern "C" void kernel_launch(void* const* d_in, const int* in_sizes, int n_in,
                              void* d_out, int out_size, void* d_ws, size_t ws_size,
                              hipStream_t stream)
{
    (void)in_sizes; (void)n_in; (void)out_size; (void)ws_size;
    const int LENS[4] = { 512, 256, 128, 64 };

    const float* x  = (const float*)d_in[0];
    const float* w1 = (const float*)d_in[1];
    const float* w2[4];  for (int l = 0; l < 4; ++l) w2[l]  = (const float*)d_in[2 + l];
    const float* saw[4]; for (int l = 0; l < 4; ++l) saw[l] = (const float*)d_in[6 + l];
    const float* sab[4]; for (int l = 0; l < 4; ++l) sab[l] = (const float*)d_in[10 + l];
    const float* lng[4]; for (int l = 0; l < 4; ++l) lng[l] = (const float*)d_in[14 + l];
    const float* lnb[4]; for (int l = 0; l < 4; ++l) lnb[l] = (const float*)d_in[18 + l];
    // mamba[l]: in_proj, conv_w, conv_b, x_proj, dt_w, dt_b, A_log, D, out_proj
    const float *m_inp[4], *m_cw[4], *m_cb[4], *m_xp[4], *m_dtw[4],
                *m_dtb[4], *m_al[4], *m_D[4], *m_op[4];
    for (int l = 0; l < 4; ++l) {
        int base = 22 + l * 9;
        m_inp[l] = (const float*)d_in[base + 0];
        m_cw[l]  = (const float*)d_in[base + 1];
        m_cb[l]  = (const float*)d_in[base + 2];
        m_xp[l]  = (const float*)d_in[base + 3];
        m_dtw[l] = (const float*)d_in[base + 4];
        m_dtb[l] = (const float*)d_in[base + 5];
        m_al[l]  = (const float*)d_in[base + 6];
        m_D[l]   = (const float*)d_in[base + 7];
        m_op[l]  = (const float*)d_in[base + 8];
    }

    float* ws = (float*)d_ws;
    size_t off = 0;
    auto alloc = [&](size_t n) { float* p = ws + off; off += n; return p; };
    float* yh[4];
    yh[0] = alloc((size_t)16 * 256 * 512);
    yh[1] = alloc((size_t)16 * 256 * 256);
    yh[2] = alloc((size_t)16 * 256 * 128);
    yh[3] = alloc((size_t)16 * 256 * 64);
    float* yl    = alloc((size_t)16 * 256 * 64);
    float* outft = alloc((size_t)16 * 256 * 64);
    float* S     = alloc((size_t)16 * 512 * 256);
    float* U     = alloc((size_t)16 * 256 * 512);
    float* XZ    = alloc((size_t)16 * 256 * 1024);
    float* XHC   = alloc((size_t)16 * 256 * 512);
    float* DBL   = alloc((size_t)16 * 256 * 64);
    float* DTF   = alloc((size_t)16 * 256 * 512);
    float* Y2    = alloc((size_t)16 * 256 * 512);
    float* MO    = alloc((size_t)16 * 256 * 512);
    float* t128  = alloc((size_t)16 * 256 * 128);
    float* t256  = alloc((size_t)16 * 256 * 256);
    float* t512  = alloc((size_t)16 * 256 * 512);

    auto launch_gemm = [&](const float* A, int aMode, int aParam,
                           const float* B, int ldb,
                           const float* bias, float* C, int ldc,
                           int M, int N, int K, int act) {
        dim3 grid((N + BN - 1) / BN, (M + BM - 1) / BM);
        gemm_wmma_kernel<<<grid, 256, 0, stream>>>(A, aMode, aParam, B, ldb,
                                                   bias, C, ldc, M, N, K, act);
    };

    // 1) 4-level DWT
    dwt_kernel<<<16 * 256, 512, 0, stream>>>(x, yh[0], yh[1], yh[2], yh[3], yl);

    // 2) out_ft = einsum('bix,iox->box', yl, weights1)
    bmm_pos_kernel<<<dim3(64 / 16, 16), 512, 0, stream>>>(yl, 0, w1, outft, 64);

    // 3) per-level detail processing
    for (int l = 0; l < 4; ++l) {
        int L = LENS[l];
        int R = (L + 15) / 16;
        int Mrows = 16 * 256;                       // mamba row count (b*s)

        // S[(b,x), o] = gelu(yh^T @ sa_w^T + sa_b)
        launch_gemm(yh[l], 1, L, saw[l], 256, sab[l], S, 256,
                    16 * L, 256, 256, ACT_GELU);
        // U[b, o, x] = einsum('bix,iox->box', S, weights2)
        bmm_pos_kernel<<<dim3(L / 16, 16), 512, 0, stream>>>(S, 1, w2[l], U, L);
        // mamba
        launch_gemm(U, 0, L, m_inp[l], L, nullptr, XZ, 2 * L,
                    Mrows, 2 * L, L, ACT_NONE);
        conv_silu_kernel<<<(16 * 256 * L + 255) / 256, 256, 0, stream>>>(
            XZ, m_cw[l], m_cb[l], XHC, L);
        launch_gemm(XHC, 0, L, m_xp[l], L, nullptr, DBL, R + 32,
                    Mrows, R + 32, L, ACT_NONE);
        launch_gemm(DBL, 0, R + 32, m_dtw[l], R, m_dtb[l], DTF, L,
                    Mrows, L, R, ACT_SOFTPLUS);
        scan_kernel<<<(16 * L + 255) / 256, 256, 0, stream>>>(
            DTF, XHC, DBL, XZ, m_al[l], m_D[l], Y2, L, R);
        launch_gemm(Y2, 0, L, m_op[l], L, nullptr, MO, L,
                    Mrows, L, L, ACT_NONE);
        // residual + layernorm (in place into yh[l] -> new_yh[l])
        ln_res_kernel<<<(16 * L + 7) / 8, 256, 0, stream>>>(
            yh[l], MO, lng[l], lnb[l], L);
    }

    // 4) inverse DWT chain
    float* outp = (float*)d_out;
    idwt_kernel<<<(16 * 256 * 64 + 255) / 256, 256, 0, stream>>>(outft, yh[3], t128, 64);
    idwt_kernel<<<(16 * 256 * 128 + 255) / 256, 256, 0, stream>>>(t128, yh[2], t256, 128);
    idwt_kernel<<<(16 * 256 * 256 + 255) / 256, 256, 0, stream>>>(t256, yh[1], t512, 256);
    idwt_kernel<<<(16 * 256 * 512 + 255) / 256, 256, 0, stream>>>(t512, yh[0], outp, 512);
}
